// MultiheadSelfAttention_12317966205513
// MI455X (gfx1250) — matrix operand order repaired and checked by reading, not compiled, not run on hardware
//
#include <hip/hip_runtime.h>
#include <hip/hip_bf16.h>
#include <type_traits>

// ---------------------------------------------------------------------------
// Multihead causal self-attention for MI455X (gfx1250, wave32, WMMA).
// All matmuls use v_wmma_f32_16x16x32_bf16 (bf16 in, fp32 accumulate).
// Flash-attention style online softmax; row-sum fused into the PV WMMA via a
// ones-column appended to V (no ds_bpermute for sums); row-max reduction via
// v_permlane16 (VALU) instead of LDS bpermute.
// Global->LDS tile movement uses GLOBAL_LOAD_ASYNC_TO_LDS_B128 (ASYNCcnt)
// with double-buffered LDS tiles. Causal masking only on diagonal blocks.
//
// Workspace layout (bytes), total 92,274,688 (~88 MB):
//   xb   @ 0         : x as bf16            [8192][1024]
//   wqb  @ 16777216  : Wq bf16 (orig layout [1024 N][1024 K])
//   wkb  @ 18874368  : Wk bf16
//   wvb  @ 20971520  : Wv bf16
//   wob  @ 23068672  : Wo bf16
//   qb   @ 25165824  : Q bf16 [8192][1024], pre-scaled by log2(e)/sqrt(dk)
//   kb   @ 41943040  : K bf16 [8192][1024]
//   vtb  @ 58720256  : V^T bf16 [64 bh][64 dk][2048 s]
//   ob   @ 75497472  : attention output bf16 [8192][1024]
// ---------------------------------------------------------------------------

typedef __attribute__((ext_vector_type(16))) __bf16 v16bf;
typedef __attribute__((ext_vector_type(8)))  __bf16 v8bf;
typedef __attribute__((ext_vector_type(8)))  float  v8f;
typedef __attribute__((ext_vector_type(4)))  int    v4i;

union V16 { v16bf v; v8bf h[2]; };

static constexpr int D_MODEL = 1024;
static constexpr int SEQ     = 2048;
static constexpr int BATCH   = 4;
static constexpr int NHEAD   = 16;
static constexpr int DKH     = 64;
static constexpr int MROWS   = BATCH * SEQ;   // 8192

#if defined(__gfx1250__) && __has_builtin(__builtin_amdgcn_global_load_async_to_lds_b128) && __has_builtin(__builtin_amdgcn_s_wait_asynccnt)
#define HAVE_ASYNC_LDS 1
#else
#define HAVE_ASYNC_LDS 0
#endif

// async-to-LDS builtin takes int4* pointers (per hipcc diagnostic);
// AS1 = global source, AS3 = LDS destination.
typedef __attribute__((address_space(1))) v4i g_v4i;
typedef __attribute__((address_space(3))) v4i l_v4i;
#define AS1V(p) ((g_v4i*)(p))
#define AS3V(p) ((l_v4i*)(p))

// ---- cross-lane max over 16-lane groups (butterfly step with xor mask) ----
#if defined(__gfx1250__) && __has_builtin(__builtin_amdgcn_permlane16)
template <int M>
__device__ inline float xor16_max(float v) {
  unsigned lo = 0, hs = 0;
#pragma unroll
  for (int j = 0; j < 8; ++j)  lo |= (unsigned)((j ^ M) & 0xF) << (4 * j);
#pragma unroll
  for (int j = 8; j < 16; ++j) hs |= (unsigned)((j ^ M) & 0xF) << (4 * (j - 8));
  int vi = __float_as_int(v);
  int pi = __builtin_amdgcn_permlane16(vi, vi, lo, hs, false, false);
  return fmaxf(v, __int_as_float(pi));
}
#else
template <int M>
__device__ inline float xor16_max(float v) {
  return fmaxf(v, __shfl_xor(v, M, 32));
}
#endif

// ------------------------------ fp32 -> bf16 -------------------------------
__global__ void cvt_f32_to_bf16(const float* __restrict__ in,
                                __bf16* __restrict__ out, int n) {
  int idx = (blockIdx.x * blockDim.x + threadIdx.x) * 8;
  if (idx + 8 > n) return;
  float4 a = *(const float4*)(in + idx);
  float4 b = *(const float4*)(in + idx + 4);
  v8bf p;
  p[0] = (__bf16)a.x; p[1] = (__bf16)a.y; p[2] = (__bf16)a.z; p[3] = (__bf16)a.w;
  p[4] = (__bf16)b.x; p[5] = (__bf16)b.y; p[6] = (__bf16)b.z; p[7] = (__bf16)b.w;
  *(v8bf*)(out + idx) = p;
}

// ------------------------------- tiled GEMM --------------------------------
// C[M,N] = A[M,K] * Bt[N,K]^T ; M-tile 128, N-tile 128, K-step 64.
// 8 waves arranged 4(M) x 2(N): each wave owns 32(M) x 64(N) = 2x4 WMMA tiles.
// Double-buffered LDS tiles, async global->LDS pipeline.
// MODE 0: bf16 out, row-major [.][1024], scaled.
// MODE 1: bf16 out, transposed V store into Vt[bh][dk][S].
// MODE 2: f32 out, row-major [.][1024].
template<int MODE>
__global__ __launch_bounds__(256) void gemm128(const __bf16* __restrict__ A,
                                               const __bf16* __restrict__ Bt,
                                               void* __restrict__ Out,
                                               float scale) {
  constexpr int LDT = 72;  // 64 + 8 pad (bank-conflict-free b128 frag reads)
  __shared__ __bf16 sA[2][128 * LDT];
  __shared__ __bf16 sB[2][128 * LDT];

  const int m0   = blockIdx.x * 128;
  const int n0   = blockIdx.y * 128;
  const int wid  = threadIdx.x >> 5;
  const int lane = threadIdx.x & 31;
  const int hi   = lane >> 4;
  const int kh   = hi * 8;      // A-frag K offset within half
  const int mwb  = (wid >> 1) * 32;
  const int nwb  = (wid & 1) * 64;

  v8f acc[2][4] = {};

  // issue one 128x64 tile pair (A and Bt) into LDS buffer `buf`
  auto issue_tile = [&](int kt, int buf) {
    for (int cc = 0; cc < 4; ++cc) {
      int c   = threadIdx.x + cc * 256;
      int row = c >> 3;
      int col = (c & 7) * 8;
#if HAVE_ASYNC_LDS
      __builtin_amdgcn_global_load_async_to_lds_b128(
          AS1V(A + (size_t)(m0 + row) * D_MODEL + kt + col),
          AS3V(&sA[buf][row * LDT + col]), 0, 0);
      __builtin_amdgcn_global_load_async_to_lds_b128(
          AS1V(Bt + (size_t)(n0 + row) * D_MODEL + kt + col),
          AS3V(&sB[buf][row * LDT + col]), 0, 0);
#else
      *(v8bf*)(&sA[buf][row * LDT + col]) =
          *(const v8bf*)(A + (size_t)(m0 + row) * D_MODEL + kt + col);
      *(v8bf*)(&sB[buf][row * LDT + col]) =
          *(const v8bf*)(Bt + (size_t)(n0 + row) * D_MODEL + kt + col);
#endif
    }
  };

  issue_tile(0, 0);

  for (int t = 0; t < D_MODEL / 64; ++t) {
#if HAVE_ASYNC_LDS
    __builtin_amdgcn_s_wait_asynccnt(0);
#endif
    __syncthreads();
    if (t + 1 < D_MODEL / 64) issue_tile((t + 1) * 64, (t + 1) & 1);

    const __bf16* tA = sA[t & 1];
    const __bf16* tB = sB[t & 1];
#pragma unroll
    for (int ks = 0; ks < 2; ++ks) {
      V16 af[2];
#pragma unroll
      for (int sm = 0; sm < 2; ++sm) {
        const __bf16* base = tA + (mwb + sm * 16 + (lane & 15)) * LDT + ks * 32;
        af[sm].h[0] = *(const v8bf*)(base + kh);
        af[sm].h[1] = *(const v8bf*)(base + 16 + kh);
      }
#pragma unroll
      for (int sn = 0; sn < 4; ++sn) {
        V16 bfrag;
        const __bf16* base =
            tB + (nwb + sn * 16 + (lane & 15)) * LDT + ks * 32 + kh;
        bfrag.h[0] = *(const v8bf*)(base);
        bfrag.h[1] = *(const v8bf*)(base + 16);
#pragma unroll
        for (int sm = 0; sm < 2; ++sm) {
          acc[sm][sn] = __builtin_amdgcn_wmma_f32_16x16x32_bf16(
              false, af[sm].v, false, bfrag.v, (short)0, acc[sm][sn],
              false, false);
        }
      }
    }
  }

  // store: C element (i) of lane -> row = gm + i, col = gn
#pragma unroll
  for (int sm = 0; sm < 2; ++sm) {
#pragma unroll
    for (int sn = 0; sn < 4; ++sn) {
      int gm = m0 + mwb + sm * 16 + hi * 8;
      int gn = n0 + nwb + sn * 16 + (lane & 15);
      v8f a = acc[sm][sn];
      if (MODE == 0) {
        __bf16* o = (__bf16*)Out;
#pragma unroll
        for (int i = 0; i < 8; ++i)
          o[(size_t)(gm + i) * D_MODEL + gn] = (__bf16)(a[i] * scale);
      } else if (MODE == 2) {
        float* o = (float*)Out;
#pragma unroll
        for (int i = 0; i < 8; ++i)
          o[(size_t)(gm + i) * D_MODEL + gn] = a[i];
      } else {  // MODE 1: transposed V store -> Vt[bh*64 + j][s]
        int b  = gm >> 11;        // / SEQ
        int s  = gm & 2047;       // within sequence (i adds consecutively)
        int bh = b * NHEAD + (gn >> 6);
        int j  = gn & 63;
        v8bf pk;
#pragma unroll
        for (int i = 0; i < 8; ++i) pk[i] = (__bf16)a[i];
        *(v8bf*)((__bf16*)Out + ((size_t)(bh * DKH + j)) * SEQ + s) = pk;
      }
    }
  }
}

// ---------------------------- flash attention ------------------------------
// grid (S/128, B*H). 8 waves x 16 query rows. 64-key blocks, causal.
// Q pre-scaled by log2(e)/sqrt(dk): softmax computed in exp2 domain.
// K/V tiles double-buffered with async global->LDS loads.
// V tile has 80 dk-rows: row 64 is all-ones so the 5th PV n-tile accumulates
// the softmax row-sum in accumulator column 64 (rescaled with O for free).
__global__ __launch_bounds__(256) void flash_attn(
    const __bf16* __restrict__ Qb, const __bf16* __restrict__ Kb,
    const __bf16* __restrict__ Vt, __bf16* __restrict__ Ob) {
  constexpr int LDT = 72;  // 64 keys + 8 pad
  __shared__ __bf16 sK[2][64 * LDT];
  __shared__ __bf16 sV[2][80 * LDT];   // rows 64..79: ones-column block
  __shared__ __bf16 sP[8][16 * LDT];

  const int tq   = blockIdx.x;
  const int bh   = blockIdx.y;
  const int b    = bh >> 4;
  const int h    = bh & 15;
  const int wid  = threadIdx.x >> 5;
  const int lane = threadIdx.x & 31;
  const int hi   = lane >> 4;
  const int kh   = hi * 8;

  // init ones-row block of both V buffers: dk-row 64 = 1.0, rows 65..79 = 0
  for (int r = threadIdx.x; r < 2 * 16 * 64; r += 256) {
    int bufr = r >> 10;          // /1024
    int rr   = (r >> 6) & 15;    // row - 64
    int cc   = r & 63;
    sV[bufr][(64 + rr) * LDT + cc] = (rr == 0) ? (__bf16)1.0f : (__bf16)0.0f;
  }

  // load Q fragments straight from global (rows strided, 16B chunks)
  const int qrow = tq * 128 + wid * 16 + (lane & 15);
  const __bf16* qp = Qb + ((size_t)(b * SEQ + qrow)) * D_MODEL + h * DKH;
  V16 qf[2];
  qf[0].h[0] = *(const v8bf*)(qp + kh);
  qf[0].h[1] = *(const v8bf*)(qp + 16 + kh);
  qf[1].h[0] = *(const v8bf*)(qp + 32 + kh);
  qf[1].h[1] = *(const v8bf*)(qp + 48 + kh);

  v8f oacc[5] = {};                    // [0..3]=O, [4] col64 = row-sum
  float mrow[8];
#pragma unroll
  for (int i = 0; i < 8; ++i) mrow[i] = -1e30f;

  const int qmin     = tq * 128 + wid * 16;   // lowest query row in wave
  const int qbase    = qmin + hi * 8;         // row slot i -> qbase+i
  const int wave_qmx = qmin + 15;
  const int nkb      = 2 * (tq + 1);

  // issue K/V tile (64 keys) for key block starting at k0 into buffer `buf`
  auto issue_kv = [&](int k0, int buf) {
    for (int cc = 0; cc < 2; ++cc) {
      int c   = threadIdx.x + cc * 256;
      int row = c >> 3;
      int col = (c & 7) * 8;
#if HAVE_ASYNC_LDS
      __builtin_amdgcn_global_load_async_to_lds_b128(
          AS1V(Kb + (size_t)(b * SEQ + k0 + row) * D_MODEL + h * DKH + col),
          AS3V(&sK[buf][row * LDT + col]), 0, 0);
      __builtin_amdgcn_global_load_async_to_lds_b128(
          AS1V(Vt + ((size_t)(bh * DKH + row)) * SEQ + k0 + col),
          AS3V(&sV[buf][row * LDT + col]), 0, 0);
#else
      *(v8bf*)(&sK[buf][row * LDT + col]) = *(const v8bf*)(
          Kb + (size_t)(b * SEQ + k0 + row) * D_MODEL + h * DKH + col);
      *(v8bf*)(&sV[buf][row * LDT + col]) =
          *(const v8bf*)(Vt + ((size_t)(bh * DKH + row)) * SEQ + k0 + col);
#endif
    }
  };

  // one key block: scores, online softmax, P*V. MASKED selects causal path.
  auto do_block = [&](int k0, const __bf16* tK, const __bf16* tV,
                      auto masked_c) {
    constexpr bool MASKED = decltype(masked_c)::value;

    // S = Q * K^T  (scores already carry log2e/sqrt(dk) via Q)
    v8f sacc[4] = {};
#pragma unroll
    for (int ks = 0; ks < 2; ++ks) {
#pragma unroll
      for (int t = 0; t < 4; ++t) {
        V16 bfrag;
        const __bf16* base =
            tK + (t * 16 + (lane & 15)) * LDT + ks * 32 + kh;
        bfrag.h[0] = *(const v8bf*)(base);
        bfrag.h[1] = *(const v8bf*)(base + 16);
        sacc[t] = __builtin_amdgcn_wmma_f32_16x16x32_bf16(
            false, qf[ks].v, false, bfrag.v, (short)0, sacc[t], false, false);
      }
    }

    // online softmax (exp2 domain); row stats replicated across half-wave
    const int keycol = k0 + (lane & 15);
#pragma unroll
    for (int i = 0; i < 8; ++i) {
      float mx = -1e30f;
#pragma unroll
      for (int t = 0; t < 4; ++t) {
        float s = sacc[t][i];
        if (MASKED && (keycol + t * 16) > (qbase + i)) s = -1e30f;
        mx = fmaxf(mx, s);
      }
      mx = xor16_max<1>(mx);
      mx = xor16_max<2>(mx);
      mx = xor16_max<4>(mx);
      mx = xor16_max<8>(mx);
      float mnew = fmaxf(mrow[i], mx);
      float sc = __builtin_amdgcn_exp2f(mrow[i] - mnew);
#pragma unroll
      for (int t = 0; t < 5; ++t) oacc[t][i] *= sc;  // rescales row-sum too
      mrow[i] = mnew;
    }

    // P = exp2(S - m) -> per-wave LDS (bf16); row-sum comes from ones-column
#pragma unroll
    for (int i = 0; i < 8; ++i) {
#pragma unroll
      for (int t = 0; t < 4; ++t) {
        float p = __builtin_amdgcn_exp2f(sacc[t][i] - mrow[i]);
        if (MASKED && (keycol + t * 16) > (qbase + i)) p = 0.f;
        sP[wid][(i + 8 * hi) * LDT + t * 16 + (lane & 15)] = (__bf16)p;
      }
    }

    // P (A-layout) from per-wave LDS; O += P * V  (t=4 -> row-sum column)
    V16 pf[2];
#pragma unroll
    for (int ks = 0; ks < 2; ++ks) {
      const __bf16* base = &sP[wid][(lane & 15) * LDT + ks * 32];
      pf[ks].h[0] = *(const v8bf*)(base + kh);
      pf[ks].h[1] = *(const v8bf*)(base + 16 + kh);
    }
#pragma unroll
    for (int t = 0; t < 5; ++t) {
#pragma unroll
      for (int ks = 0; ks < 2; ++ks) {
        V16 vfrag;
        const __bf16* base =
            tV + (t * 16 + (lane & 15)) * LDT + ks * 32 + kh;
        vfrag.h[0] = *(const v8bf*)(base);
        vfrag.h[1] = *(const v8bf*)(base + 16);
        oacc[t] = __builtin_amdgcn_wmma_f32_16x16x32_bf16(
            false, pf[ks].v, false, vfrag.v, (short)0, oacc[t], false, false);
      }
    }
  };

  issue_kv(0, 0);

  for (int kb = 0; kb < nkb; ++kb) {
    const int k0 = kb * 64;
#if HAVE_ASYNC_LDS
    __builtin_amdgcn_s_wait_asynccnt(0);
#endif
    __syncthreads();
    if (kb + 1 < nkb) issue_kv((kb + 1) * 64, (kb + 1) & 1);

    if (k0 <= wave_qmx) {  // else fully masked for this wave's rows
      const __bf16* tK = sK[kb & 1];
      const __bf16* tV = sV[kb & 1];
      if (k0 + 63 <= qmin)  // interior block: no causal masking needed
        do_block(k0, tK, tV, std::integral_constant<bool, false>{});
      else                  // diagonal block: per-element causal mask
        do_block(k0, tK, tV, std::integral_constant<bool, true>{});
    }
  }

  // row-sum lives in accumulator column 64 (lane 0 / 16 of each half-group)
  float inv[8];
#pragma unroll
  for (int i = 0; i < 8; ++i) {
    float s = __shfl(oacc[4][i], hi * 16, 32);  // broadcast within half
    inv[i] = 1.0f / s;
  }
#pragma unroll
  for (int t = 0; t < 4; ++t) {
    int col = h * DKH + t * 16 + (lane & 15);
#pragma unroll
    for (int i = 0; i < 8; ++i) {
      int q = qbase + i;
      float o = oacc[t][i] * inv[i];
      Ob[((size_t)(b * SEQ + q)) * D_MODEL + col] = (__bf16)o;
    }
  }
}

// ------------------------------- launcher ----------------------------------
extern "C" void kernel_launch(void* const* d_in, const int* in_sizes, int n_in,
                              void* d_out, int out_size, void* d_ws,
                              size_t ws_size, hipStream_t stream) {
  const float* x  = (const float*)d_in[0];
  const float* Wq = (const float*)d_in[1];
  const float* Wk = (const float*)d_in[2];
  const float* Wv = (const float*)d_in[3];
  const float* Wo = (const float*)d_in[4];

  char* ws = (char*)d_ws;
  __bf16* xb  = (__bf16*)(ws + 0);
  __bf16* wqb = (__bf16*)(ws + 16777216);
  __bf16* wkb = (__bf16*)(ws + 18874368);
  __bf16* wvb = (__bf16*)(ws + 20971520);
  __bf16* wob = (__bf16*)(ws + 23068672);
  __bf16* qb  = (__bf16*)(ws + 25165824);
  __bf16* kb  = (__bf16*)(ws + 41943040);
  __bf16* vtb = (__bf16*)(ws + 58720256);
  __bf16* ob  = (__bf16*)(ws + 75497472);

  // fp32 -> bf16 conversions (2048 elems per block)
  cvt_f32_to_bf16<<<MROWS * D_MODEL / 2048, 256, 0, stream>>>(x, xb, MROWS * D_MODEL);
  cvt_f32_to_bf16<<<D_MODEL * D_MODEL / 2048, 256, 0, stream>>>(Wq, wqb, D_MODEL * D_MODEL);
  cvt_f32_to_bf16<<<D_MODEL * D_MODEL / 2048, 256, 0, stream>>>(Wk, wkb, D_MODEL * D_MODEL);
  cvt_f32_to_bf16<<<D_MODEL * D_MODEL / 2048, 256, 0, stream>>>(Wv, wvb, D_MODEL * D_MODEL);
  cvt_f32_to_bf16<<<D_MODEL * D_MODEL / 2048, 256, 0, stream>>>(Wo, wob, D_MODEL * D_MODEL);

  // QKV projections; Q pre-scaled by log2(e)/sqrt(dk) for exp2-domain softmax
  const float qscale = 0.125f * 1.4426950408889634f;
  dim3 gproj(MROWS / 128, D_MODEL / 128);
  gemm128<0><<<gproj, 256, 0, stream>>>(xb, wqb, qb, qscale);
  gemm128<0><<<gproj, 256, 0, stream>>>(xb, wkb, kb, 1.0f);
  gemm128<1><<<gproj, 256, 0, stream>>>(xb, wvb, vtb, 1.0f);

  // causal flash attention
  dim3 gattn(SEQ / 128, BATCH * NHEAD);
  flash_attn<<<gattn, 256, 0, stream>>>(qb, kb, vtb, ob);

  // output projection -> fp32 d_out
  gemm128<2><<<gproj, 256, 0, stream>>>(ob, wob, (float*)d_out, 1.0f);
}
